// G_HANet_Surv_44710609552061
// MI455X (gfx1250) — matrix operand order repaired and checked by reading, not compiled, not run on hardware
//
#include <hip/hip_runtime.h>
#include <cstddef>
#include <cstdint>

#define NPATCH 16384
#define KTOP   3276u
#define TT     6
#define HH     4
#define DHD    128

typedef __attribute__((ext_vector_type(16))) __bf16 v16bf;
typedef __attribute__((ext_vector_type(8)))  __bf16 v8bf;
typedef __attribute__((ext_vector_type(8)))  float  v8f;

union BfFrag { v16bf v; v8bf h[2]; };

__device__ __forceinline__ __bf16 to_bf16(float f) {
  union { float f; unsigned int u; } a; a.f = f;
  unsigned int u = a.u;
  u += 0x7fffu + ((u >> 16) & 1u);           // round-to-nearest-even
  union { unsigned short s; __bf16 b; } r; r.s = (unsigned short)(u >> 16);
  return r.b;
}
__device__ __forceinline__ float bf_to_f(__bf16 x) {
  union { __bf16 b; unsigned short s; } u; u.b = x;
  union { unsigned int i; float f; } r; r.i = ((unsigned int)u.s) << 16;
  return r.f;
}

// ---------------- conversion kernels ----------------
__global__ void cvt_bf16(const float* __restrict__ src, __bf16* __restrict__ dst, int n) {
  int i = blockIdx.x * blockDim.x + threadIdx.x;
  if (i < n) dst[i] = to_bf16(src[i]);
}
// src is (K x M) row-major f32 -> dst is (M x K) row-major bf16
__global__ void transpose_cvt(const float* __restrict__ src, __bf16* __restrict__ dst,
                              int K, int M) {
  int i = blockIdx.x * blockDim.x + threadIdx.x;
  if (i < K * M) {
    int k = i / M, m = i % M;
    dst[(size_t)m * K + k] = to_bf16(src[i]);
  }
}

// ---------------- WMMA bf16 GEMM with async LDS double-buffering ----------------
// C[16384 x M] = act( A[16384 x K] * Bt[M x K]^T + bias )
// grid: (16384/128, M/64), block: 256 (8 waves). Wave w -> rows base+16w, 4x16 col tiles.
// B panel (64 cols x 128 K) is staged global->LDS with GLOBAL_LOAD_ASYNC_TO_LDS_B128
// (ASYNCcnt), double-buffered; WMMAs read B fragments from LDS (DS path).
#define BROW 136   // 128 K elems + 8 pad -> 272B row stride (breaks bank-conflict stride)

__device__ __forceinline__ void stage_b(unsigned ldsBase, unsigned long long gbase,
                                        int colBase, int K, int kc, int tid) {
  // 64 rows x 128 bf16 = 1024 x 16B segments; 256 threads x 4 segments
#pragma unroll
  for (int s = 0; s < 4; ++s) {
    int seg = tid + s * 256;
    int c   = seg >> 4;          // 0..63
    int sg  = seg & 15;          // 0..15
    unsigned lds  = ldsBase + (unsigned)(c * (BROW * 2) + sg * 16);
    unsigned voff = (unsigned)((((colBase + c) * K) + kc + sg * 8) * 2);
    asm volatile("global_load_async_to_lds_b128 %0, %1, %2"
                 :: "v"(lds), "v"(voff), "s"(gbase) : "memory");
  }
}

__global__ __launch_bounds__(256)
void wmma_gemm_bf16(const __bf16* __restrict__ A, const __bf16* __restrict__ Bt,
                    const float* __restrict__ bias,
                    float* __restrict__ Cf, __bf16* __restrict__ Cb,
                    int K, int M, int act) {
  __shared__ __attribute__((aligned(16))) __bf16 btile[2][64][BROW];

  const int tid     = threadIdx.x;
  const int lane    = tid & 31;
  const int wave    = tid >> 5;
  const int rowBase = blockIdx.x * 128 + wave * 16;
  const int colBase = blockIdx.y * 64;
  const int l15     = lane & 15;
  const int hs      = lane >> 4;          // half-wave select
  const int bOff    = hs ? 8 : 0;

  v8f zero = {};
  v8f acc[4] = {zero, zero, zero, zero};

  const __bf16* aRow = A + (size_t)(rowBase + l15) * K + (hs ? 8 : 0);
  const unsigned long long gB = (unsigned long long)(uintptr_t)Bt;
  const unsigned ldsB0 = (unsigned)(uintptr_t)&btile[0][0][0];
  const unsigned ldsB1 = (unsigned)(uintptr_t)&btile[1][0][0];

  const int nch = K >> 7;                 // K / 128 (K is 512 or 1024)
  stage_b(ldsB0, gB, colBase, K, 0, tid);

  for (int ch = 0; ch < nch; ++ch) {
    const int kc = ch << 7;
    if (ch + 1 < nch) {
      stage_b((ch & 1) ? ldsB0 : ldsB1, gB, colBase, K, kc + 128, tid);
      asm volatile("s_wait_asynccnt 0x4" ::: "memory");  // current chunk done, next in flight
    } else {
      asm volatile("s_wait_asynccnt 0x0" ::: "memory");
    }
    __syncthreads();

    const __bf16* bt = &btile[ch & 1][0][0];
#pragma unroll
    for (int ks = 0; ks < 4; ++ks) {
      const int kb = kc + ks * 32;
      BfFrag afr;
      afr.h[0] = *(const v8bf*)(aRow + kb);
      afr.h[1] = *(const v8bf*)(aRow + kb + 16);
      const int kk0 = ks * 32 + bOff;
#pragma unroll
      for (int j = 0; j < 4; ++j) {
        const __bf16* brow = bt + (16 * j + l15) * BROW + kk0;
        BfFrag bfr;
        bfr.h[0] = *(const v8bf*)(brow);
        bfr.h[1] = *(const v8bf*)(brow + 16);
        acc[j] = __builtin_amdgcn_wmma_f32_16x16x32_bf16(
            false, afr.v, false, bfr.v, (short)0, acc[j], false, false);
      }
    }
    __syncthreads();   // all waves done reading this buffer before it is re-staged
  }

#pragma unroll
  for (int j = 0; j < 4; ++j) {
    int c = colBase + 16 * j + l15;
    float bv = bias ? bias[c] : 0.0f;
#pragma unroll
    for (int i = 0; i < 8; ++i) {
      int row = rowBase + (hs ? (i + 8) : i);
      float v = acc[j][i] + bv;
      if (act == 1) v = fmaxf(v, 0.0f);
      if (Cf) Cf[(size_t)row * M + c] = v;
      if (Cb) Cb[(size_t)row * M + c] = to_bf16(v);
    }
  }
}

// ---------------- LayerNorm over rows ----------------
__global__ void ln_rows(const float* __restrict__ X, const float* __restrict__ g,
                        const float* __restrict__ b, float* outF, __bf16* outB,
                        int cols, int do_relu) {
  int row = blockIdx.x, tid = threadIdx.x;
  __shared__ float r1[256], r2[256];
  const float* x = X + (size_t)row * cols;
  float s = 0.f, s2 = 0.f;
  for (int c = tid; c < cols; c += 256) { float v = x[c]; s += v; s2 += v * v; }
  r1[tid] = s; r2[tid] = s2; __syncthreads();
  for (int st = 128; st > 0; st >>= 1) {
    if (tid < st) { r1[tid] += r1[tid + st]; r2[tid] += r2[tid + st]; }
    __syncthreads();
  }
  float mean = r1[0] / cols;
  float var  = r2[0] / cols - mean * mean;
  float inv  = rsqrtf(var + 1e-5f);
  for (int c = tid; c < cols; c += 256) {
    float v = (x[c] - mean) * inv * g[c] + b[c];
    if (do_relu) v = fmaxf(v, 0.0f);
    if (outF) outF[(size_t)row * cols + c] = v;
    if (outB) outB[(size_t)row * cols + c] = to_bf16(v);
  }
}

// ---------------- gated attention weight per patch ----------------
__global__ void gate_weights(const float* __restrict__ a_pre, const float* __restrict__ b_pre,
                             const float* __restrict__ Wc, const float* __restrict__ bc,
                             float* __restrict__ wout) {
  int n = blockIdx.x, tid = threadIdx.x;
  __shared__ float red[256];
  float ta = tanhf(a_pre[(size_t)n * 256 + tid]);
  float sb = 1.0f / (1.0f + expf(-b_pre[(size_t)n * 256 + tid]));
  red[tid] = ta * sb * Wc[tid];
  __syncthreads();
  for (int st = 128; st > 0; st >>= 1) {
    if (tid < st) red[tid] += red[tid + st];
    __syncthreads();
  }
  if (tid == 0) wout[n] = red[0] + bc[0];
}

// ---------------- 1D softmax (edge1) ----------------
__global__ void softmax1d(const float* __restrict__ x, float* __restrict__ y, int n) {
  int tid = threadIdx.x;
  __shared__ float red[256];
  float m = -1e30f;
  for (int i = tid; i < n; i += 256) m = fmaxf(m, x[i]);
  red[tid] = m; __syncthreads();
  for (int st = 128; st > 0; st >>= 1) { if (tid < st) red[tid] = fmaxf(red[tid], red[tid + st]); __syncthreads(); }
  float mx = red[0]; __syncthreads();
  float s = 0.f;
  for (int i = tid; i < n; i += 256) s += expf(x[i] - mx);
  red[tid] = s; __syncthreads();
  for (int st = 128; st > 0; st >>= 1) { if (tid < st) red[tid] += red[tid + st]; __syncthreads(); }
  float sum = red[0];
  for (int i = tid; i < n; i += 256) y[i] = expf(x[i] - mx) / sum;
}

// ---------------- small plain GEMM for token-sized math ----------------
// out[M x N] = act(A[M x K] @ W[K x N] + bias) + resid ; act: 0 none, 1 gelu(tanh)
__global__ void small_mm(const float* __restrict__ A, const float* __restrict__ W,
                         const float* __restrict__ bias, const float* __restrict__ resid,
                         float* __restrict__ out, int M, int K, int N, int act) {
  for (int id = blockIdx.x * blockDim.x + threadIdx.x; id < M * N;
       id += gridDim.x * blockDim.x) {
    int m = id / N, n = id % N;
    float a = 0.f;
    for (int k = 0; k < K; ++k) a += A[(size_t)m * K + k] * W[(size_t)k * N + n];
    if (bias) a += bias[n];
    if (act == 1) {
      float x = a;
      a = 0.5f * x * (1.0f + tanhf(0.7978845608028654f * (x + 0.044715f * x * x * x)));
    }
    if (resid) a += resid[id];
    out[id] = a;
  }
}

// ---------------- cross-attention scores (H,T,N) ----------------
__global__ void scores_k(const float* __restrict__ q, const __bf16* __restrict__ kmat,
                         float* __restrict__ attn) {
  int ht = blockIdx.y;                       // h*6+t
  int n  = blockIdx.x * blockDim.x + threadIdx.x;
  int h = ht / TT, t = ht % TT;
  const float*  qr = q + (size_t)t * 512 + h * DHD;
  const __bf16* kr = kmat + (size_t)n * 512 + h * DHD;
  float a = 0.f;
  for (int d = 0; d < DHD; ++d) a += qr[d] * bf_to_f(kr[d]);
  attn[(size_t)ht * NPATCH + n] = a * 0.08838834764831845f;  // 1/sqrt(128)
}

// in-place softmax per row of length NPATCH
__global__ void softmax_rows(float* __restrict__ attn, int n) {
  float* row = attn + (size_t)blockIdx.x * n;
  int tid = threadIdx.x;
  __shared__ float red[256];
  float m = -1e30f;
  for (int i = tid; i < n; i += 256) m = fmaxf(m, row[i]);
  red[tid] = m; __syncthreads();
  for (int st = 128; st > 0; st >>= 1) { if (tid < st) red[tid] = fmaxf(red[tid], red[tid + st]); __syncthreads(); }
  float mx = red[0]; __syncthreads();
  float s = 0.f;
  for (int i = tid; i < n; i += 256) s += expf(row[i] - mx);
  red[tid] = s; __syncthreads();
  for (int st = 128; st > 0; st >>= 1) { if (tid < st) red[tid] += red[tid + st]; __syncthreads(); }
  float sum = red[0];
  for (int i = tid; i < n; i += 256) row[i] = expf(row[i] - mx) / sum;
}

__global__ void matrix_mean(const float* __restrict__ attn, float* __restrict__ mat) {
  int id = blockIdx.x * blockDim.x + threadIdx.x;
  if (id < TT * NPATCH) {
    int t = id / NPATCH, n = id % NPATCH;
    float a = 0.f;
    for (int h = 0; h < HH; ++h) a += attn[(size_t)(h * TT + t) * NPATCH + n];
    mat[id] = a * 0.25f;
  }
}

// exact k-th largest per row via binary search on (non-negative) float bits
__global__ void topk_thresh(const float* __restrict__ mat, float* __restrict__ thresh) {
  int t = blockIdx.x, tid = threadIdx.x;
  __shared__ unsigned int red[256];
  const unsigned int* row = (const unsigned int*)(mat + (size_t)t * NPATCH);
  unsigned int lo = 0u, hi = 0x7f7fffffu;
  for (int it = 0; it < 31; ++it) {
    unsigned int mid = lo + ((hi - lo + 1u) >> 1);
    unsigned int cnt = 0;
    for (int n = tid; n < NPATCH; n += 256) cnt += (row[n] >= mid) ? 1u : 0u;
    red[tid] = cnt; __syncthreads();
    for (int st = 128; st > 0; st >>= 1) { if (tid < st) red[tid] += red[tid + st]; __syncthreads(); }
    unsigned int total = red[0]; __syncthreads();
    if (total >= KTOP) lo = mid; else hi = mid - 1u;
  }
  if (tid == 0) { union { unsigned int u; float f; } c; c.u = lo; thresh[t] = c.f; }
}

// masked softmax (edge2) and G = (edge1 + edge2)/2
__global__ void edge2G(const float* __restrict__ mat, const float* __restrict__ thresh,
                       const float* __restrict__ edge1, float* __restrict__ G) {
  int t = blockIdx.x, tid = threadIdx.x;
  __shared__ float red[256];
  const float* row = mat + (size_t)t * NPATCH;
  float th = thresh[t];
  float m = -1e30f;
  for (int n = tid; n < NPATCH; n += 256) {
    float v = row[n]; if (v < th) v -= 100000.0f;
    m = fmaxf(m, v);
  }
  red[tid] = m; __syncthreads();
  for (int st = 128; st > 0; st >>= 1) { if (tid < st) red[tid] = fmaxf(red[tid], red[tid + st]); __syncthreads(); }
  float mx = red[0]; __syncthreads();
  float s = 0.f;
  for (int n = tid; n < NPATCH; n += 256) {
    float v = row[n]; if (v < th) v -= 100000.0f;
    s += expf(v - mx);
  }
  red[tid] = s; __syncthreads();
  for (int st = 128; st > 0; st >>= 1) { if (tid < st) red[tid] += red[tid + st]; __syncthreads(); }
  float sum = red[0];
  for (int n = tid; n < NPATCH; n += 256) {
    float v = row[n]; if (v < th) v -= 100000.0f;
    G[(size_t)t * NPATCH + n] = 0.5f * (edge1[n] + expf(v - mx) / sum);
  }
}

// tok_pre[t, h*128+d] = sum_n attn[h,t,n] * v[n, h*128+d]
__global__ void tokpre_k(const float* __restrict__ attn, const __bf16* __restrict__ vmat,
                         float* __restrict__ tok_pre) {
  int t = blockIdx.x / HH, h = blockIdx.x % HH, d = threadIdx.x;
  const float* arow = attn + (size_t)(h * TT + t) * NPATCH;
  float a = 0.f;
  for (int n = 0; n < NPATCH; ++n)
    a += arow[n] * bf_to_f(vmat[(size_t)n * 512 + h * DHD + d]);
  tok_pre[(size_t)t * 512 + h * DHD + d] = a;
}

// fea[t,c] = 0.5*(relu(sum_n G[t,n]*hW[n,c]) + tok[t,c])
__global__ void feah_k(const float* __restrict__ G, const float* __restrict__ hW,
                       const float* __restrict__ tokv, float* __restrict__ fea) {
  int t = blockIdx.x >> 1;
  int c = (blockIdx.x & 1) * 256 + threadIdx.x;
  const float* g = G + (size_t)t * NPATCH;
  float a = 0.f;
  for (int n = 0; n < NPATCH; ++n) a += g[n] * hW[(size_t)n * 512 + c];
  fea[(size_t)t * 512 + c] = 0.5f * (fmaxf(a, 0.f) + tokv[(size_t)t * 512 + c]);
}

// omic_raw[t,o] = sum_d tok[t,d]*omic_W[t,d,o] + omic_b[t,o]
__global__ void omic_k(const float* __restrict__ tok, const float* __restrict__ omW,
                       const float* __restrict__ omb, float* __restrict__ omic_raw) {
  int id = blockIdx.x * blockDim.x + threadIdx.x;
  if (id < TT * 600) {
    int t = id / 600, o = id % 600;
    float a = 0.f;
    for (int d = 0; d < 512; ++d) a += tok[(size_t)t * 512 + d] * omW[((size_t)t * 512 + d) * 600 + o];
    omic_raw[id] = a + omb[id];
  }
}

// 6-token multihead self-attention core (single block)
__global__ void tok_attn(const float* __restrict__ q2, const float* __restrict__ k2,
                         const float* __restrict__ v2, float* __restrict__ sa_pre) {
  __shared__ float pr[HH][TT][TT];
  int tid = threadIdx.x;
  if (tid < HH * TT * TT) {
    int h = tid / (TT * TT), qq = (tid / TT) % TT, kk = tid % TT;
    float a = 0.f;
    for (int d = 0; d < DHD; ++d)
      a += q2[(size_t)qq * 512 + h * DHD + d] * k2[(size_t)kk * 512 + h * DHD + d];
    pr[h][qq][kk] = a * 0.08838834764831845f;
  }
  __syncthreads();
  if (tid < HH * TT) {
    int h = tid / TT, qq = tid % TT;
    float mx = -1e30f;
    for (int kk = 0; kk < TT; ++kk) mx = fmaxf(mx, pr[h][qq][kk]);
    float s = 0.f;
    for (int kk = 0; kk < TT; ++kk) { float e = expf(pr[h][qq][kk] - mx); pr[h][qq][kk] = e; s += e; }
    for (int kk = 0; kk < TT; ++kk) pr[h][qq][kk] /= s;
  }
  __syncthreads();
  for (int id = tid; id < TT * 512; id += 256) {
    int qq = id / 512, hd = id % 512, h = hd >> 7;
    float a = 0.f;
    for (int kk = 0; kk < TT; ++kk) a += pr[h][qq][kk] * v2[(size_t)kk * 512 + hd];
    sa_pre[id] = a;
  }
}

// survival head: logits -> hazards, S, Y_hat -> d_out[0..8]
__global__ void head_k(const float* __restrict__ hb, const float* __restrict__ clsW,
                       const float* __restrict__ clsb, float* __restrict__ out) {
  __shared__ float lg[4];
  int tid = threadIdx.x;
  if (tid < 4) {
    float a = 0.f;
    for (int i = 0; i < TT * 256; ++i) a += hb[i] * clsW[(size_t)i * 4 + tid];
    lg[tid] = a + clsb[tid];
  }
  __syncthreads();
  if (tid == 0) {
    int best = 0;
    for (int j = 1; j < 4; ++j) if (lg[j] > lg[best]) best = j;
    float run = 1.f;
    for (int j = 0; j < 4; ++j) {
      float hz = 1.0f / (1.0f + expf(-lg[j]));
      run *= (1.0f - hz);
      out[j] = hz;
      out[4 + j] = run;
    }
    out[8] = (float)best;
  }
}

// pack re_omics (sizes 100..600) and matrix (6x16384) into d_out
__global__ void pack_out(const float* __restrict__ omic_raw, const float* __restrict__ mat,
                         float* __restrict__ out) {
  int id = blockIdx.x * blockDim.x + threadIdx.x;
  if (id < 2100) {
    int t, o;
    if      (id < 100)  { t = 0; o = id; }
    else if (id < 300)  { t = 1; o = id - 100; }
    else if (id < 600)  { t = 2; o = id - 300; }
    else if (id < 1000) { t = 3; o = id - 600; }
    else if (id < 1500) { t = 4; o = id - 1000; }
    else                { t = 5; o = id - 1500; }
    out[9 + id] = omic_raw[t * 600 + o];
  } else if (id < 2100 + TT * NPATCH) {
    out[9 + id] = mat[id - 2100];
  }
}

extern "C" void kernel_launch(void* const* d_in, const int* in_sizes, int n_in,
                              void* d_out, int out_size, void* d_ws, size_t ws_size,
                              hipStream_t stream) {
  (void)in_sizes; (void)n_in; (void)out_size; (void)ws_size;
  const float* x_path   = (const float*)d_in[0];
  const float* wsi_W    = (const float*)d_in[1];
  const float* wsi_b    = (const float*)d_in[2];
  const float* wsi_ln_g = (const float*)d_in[3];
  const float* wsi_ln_b = (const float*)d_in[4];
  const float* gate_Wa  = (const float*)d_in[5];
  const float* gate_ba  = (const float*)d_in[6];
  const float* gate_Wb  = (const float*)d_in[7];
  const float* gate_bb  = (const float*)d_in[8];
  const float* gate_Wc  = (const float*)d_in[9];
  const float* gate_bc  = (const float*)d_in[10];
  const float* rec_W    = (const float*)d_in[11];
  const float* rec_b    = (const float*)d_in[12];
  const float* tokens   = (const float*)d_in[13];
  const float* rec_Wq   = (const float*)d_in[14];
  const float* rec_Wk   = (const float*)d_in[15];
  const float* rec_Wv   = (const float*)d_in[16];
  const float* rec_Wo   = (const float*)d_in[17];
  const float* rec_bo   = (const float*)d_in[18];
  const float* omic_W   = (const float*)d_in[19];
  const float* omic_b   = (const float*)d_in[20];
  const float* hyp_W    = (const float*)d_in[21];
  const float* hyp_b    = (const float*)d_in[22];
  const float* att_Wq   = (const float*)d_in[23];
  const float* att_Wk   = (const float*)d_in[24];
  const float* att_Wv   = (const float*)d_in[25];
  const float* att_Wo   = (const float*)d_in[26];
  const float* att_bo   = (const float*)d_in[27];
  const float* ln1_g    = (const float*)d_in[28];
  const float* ln1_b    = (const float*)d_in[29];
  const float* ffn_W1   = (const float*)d_in[30];
  const float* ffn_b1   = (const float*)d_in[31];
  const float* ffn_W2   = (const float*)d_in[32];
  const float* ffn_b2   = (const float*)d_in[33];
  const float* but_W    = (const float*)d_in[34];
  const float* but_b    = (const float*)d_in[35];
  const float* but_ln_g = (const float*)d_in[36];
  const float* but_ln_b = (const float*)d_in[37];
  const float* cls_W    = (const float*)d_in[38];
  const float* cls_b    = (const float*)d_in[39];
  float* out = (float*)d_out;

  char* base = (char*)d_ws;
  size_t off = 0;
  auto alloc = [&](size_t bytes) -> void* {
    void* p = base + off;
    off += (bytes + 255) & ~(size_t)255;
    return p;
  };
  const size_t N = NPATCH;
  __bf16* xb      = (__bf16*)alloc(N * 1024 * 2);
  __bf16* recW_t  = (__bf16*)alloc(512 * 1024 * 2);
  __bf16* wsiW_t  = (__bf16*)alloc(512 * 1024 * 2);
  __bf16* recWk_t = (__bf16*)alloc(512 * 512 * 2);
  __bf16* recWv_t = (__bf16*)alloc(512 * 512 * 2);
  __bf16* gWa_t   = (__bf16*)alloc(256 * 512 * 2);
  __bf16* gWb_t   = (__bf16*)alloc(256 * 512 * 2);
  __bf16* hypW_t  = (__bf16*)alloc(512 * 512 * 2);
  __bf16* hp_b    = (__bf16*)alloc(N * 512 * 2);
  __bf16* k_b     = (__bf16*)alloc(N * 512 * 2);
  __bf16* v_b     = (__bf16*)alloc(N * 512 * 2);
  float*  hpre    = (float*)alloc(N * 512 * 4);   // later reused as hW
  __bf16* h_b     = (__bf16*)alloc(N * 512 * 2);
  float*  a_pre   = (float*)alloc(N * 256 * 4);
  float*  b_pre   = (float*)alloc(N * 256 * 4);
  float*  wvec    = (float*)alloc(N * 4);
  float*  edge1   = (float*)alloc(N * 4);
  float*  qbuf    = (float*)alloc(TT * 512 * 4);
  float*  attn    = (float*)alloc((size_t)HH * TT * N * 4);
  float*  matbuf  = (float*)alloc((size_t)TT * N * 4);
  float*  thresh  = (float*)alloc(TT * 4);
  float*  Gm      = (float*)alloc((size_t)TT * N * 4);
  float*  tok_pre = (float*)alloc(TT * 512 * 4);
  float*  tokv    = (float*)alloc(TT * 512 * 4);
  float*  omicraw = (float*)alloc(TT * 600 * 4);
  float*  fea     = (float*)alloc(TT * 512 * 4);
  float*  q2      = (float*)alloc(TT * 512 * 4);
  float*  k2      = (float*)alloc(TT * 512 * 4);
  float*  v2      = (float*)alloc(TT * 512 * 4);
  float*  sa_pre  = (float*)alloc(TT * 512 * 4);
  float*  fea2    = (float*)alloc(TT * 512 * 4);
  float*  x1      = (float*)alloc(TT * 512 * 4);
  float*  ffn1    = (float*)alloc(TT * 2048 * 4);
  float*  fea3    = (float*)alloc(TT * 512 * 4);
  float*  hb_pre  = (float*)alloc(TT * 256 * 4);
  float*  hb      = (float*)alloc(TT * 256 * 4);

  // ---- stage 0: conversions ----
  cvt_bf16<<<(int)((N * 1024 + 255) / 256), 256, 0, stream>>>(x_path, xb, (int)(N * 1024));
  transpose_cvt<<<(1024 * 512 + 255) / 256, 256, 0, stream>>>(rec_W,  recW_t,  1024, 512);
  transpose_cvt<<<(1024 * 512 + 255) / 256, 256, 0, stream>>>(wsi_W,  wsiW_t,  1024, 512);
  transpose_cvt<<<(512 * 512 + 255) / 256, 256, 0, stream>>>(rec_Wk, recWk_t, 512, 512);
  transpose_cvt<<<(512 * 512 + 255) / 256, 256, 0, stream>>>(rec_Wv, recWv_t, 512, 512);
  transpose_cvt<<<(512 * 256 + 255) / 256, 256, 0, stream>>>(gate_Wa, gWa_t,  512, 256);
  transpose_cvt<<<(512 * 256 + 255) / 256, 256, 0, stream>>>(gate_Wb, gWb_t,  512, 256);
  transpose_cvt<<<(512 * 512 + 255) / 256, 256, 0, stream>>>(hyp_W,  hypW_t,  512, 512);

  // ---- stage 1: big WMMA GEMMs ----
  dim3 g512(NPATCH / 128, 512 / 64), g256(NPATCH / 128, 256 / 64);
  // hp = relu(x @ rec_W + rec_b)  -> bf16
  wmma_gemm_bf16<<<g512, 256, 0, stream>>>(xb, recW_t, rec_b, nullptr, hp_b, 1024, 512, 1);
  // h_pre = x @ wsi_W + wsi_b     -> f32
  wmma_gemm_bf16<<<g512, 256, 0, stream>>>(xb, wsiW_t, wsi_b, hpre, nullptr, 1024, 512, 0);
  // k = hp @ rec_Wk ; v = hp @ rec_Wv -> bf16
  wmma_gemm_bf16<<<g512, 256, 0, stream>>>(hp_b, recWk_t, nullptr, nullptr, k_b, 512, 512, 0);
  wmma_gemm_bf16<<<g512, 256, 0, stream>>>(hp_b, recWv_t, nullptr, nullptr, v_b, 512, 512, 0);
  // h = relu(LN(h_pre)) -> bf16
  ln_rows<<<NPATCH, 256, 0, stream>>>(hpre, wsi_ln_g, wsi_ln_b, nullptr, h_b, 512, 1);
  // gates and hyperedge projection
  wmma_gemm_bf16<<<g256, 256, 0, stream>>>(h_b, gWa_t, gate_ba, a_pre, nullptr, 512, 256, 0);
  wmma_gemm_bf16<<<g256, 256, 0, stream>>>(h_b, gWb_t, gate_bb, b_pre, nullptr, 512, 256, 0);
  wmma_gemm_bf16<<<g512, 256, 0, stream>>>(h_b, hypW_t, hyp_b, hpre /*hW*/, nullptr, 512, 512, 0);

  // ---- stage 2: gated attention + cross attention ----
  gate_weights<<<NPATCH, 256, 0, stream>>>(a_pre, b_pre, gate_Wc, gate_bc, wvec);
  softmax1d<<<1, 256, 0, stream>>>(wvec, edge1, NPATCH);
  small_mm<<<12, 256, 0, stream>>>(tokens, rec_Wq, nullptr, nullptr, qbuf, TT, 512, 512, 0);
  scores_k<<<dim3(NPATCH / 256, HH * TT), 256, 0, stream>>>(qbuf, k_b, attn);
  softmax_rows<<<HH * TT, 256, 0, stream>>>(attn, NPATCH);
  matrix_mean<<<(TT * NPATCH + 255) / 256, 256, 0, stream>>>(attn, matbuf);
  topk_thresh<<<TT, 256, 0, stream>>>(matbuf, thresh);
  edge2G<<<TT, 256, 0, stream>>>(matbuf, thresh, edge1, Gm);
  tokpre_k<<<TT * HH, DHD, 0, stream>>>(attn, v_b, tok_pre);
  small_mm<<<12, 256, 0, stream>>>(tok_pre, rec_Wo, rec_bo, nullptr, tokv, TT, 512, 512, 0);
  omic_k<<<(TT * 600 + 255) / 256, 256, 0, stream>>>(tokv, omic_W, omic_b, omicraw);
  feah_k<<<TT * 2, 256, 0, stream>>>(Gm, hpre /*hW*/, tokv, fea);

  // ---- stage 3: token transformer + head ----
  small_mm<<<12, 256, 0, stream>>>(fea, att_Wq, nullptr, nullptr, q2, TT, 512, 512, 0);
  small_mm<<<12, 256, 0, stream>>>(fea, att_Wk, nullptr, nullptr, k2, TT, 512, 512, 0);
  small_mm<<<12, 256, 0, stream>>>(fea, att_Wv, nullptr, nullptr, v2, TT, 512, 512, 0);
  tok_attn<<<1, 256, 0, stream>>>(q2, k2, v2, sa_pre);
  small_mm<<<12, 256, 0, stream>>>(sa_pre, att_Wo, att_bo, fea, fea2, TT, 512, 512, 0);
  ln_rows<<<TT, 256, 0, stream>>>(fea2, ln1_g, ln1_b, x1, nullptr, 512, 0);
  small_mm<<<48, 256, 0, stream>>>(x1, ffn_W1, ffn_b1, nullptr, ffn1, TT, 512, 2048, 1);
  small_mm<<<12, 256, 0, stream>>>(ffn1, ffn_W2, ffn_b2, fea2, fea3, TT, 2048, 512, 0);
  small_mm<<<6, 256, 0, stream>>>(fea3, but_W, but_b, nullptr, hb_pre, TT, 512, 256, 0);
  ln_rows<<<TT, 256, 0, stream>>>(hb_pre, but_ln_g, but_ln_b, hb, nullptr, 256, 1);
  head_k<<<1, 64, 0, stream>>>(hb, cls_W, cls_b, out);
  pack_out<<<(2100 + TT * NPATCH + 255) / 256, 256, 0, stream>>>(omicraw, matbuf, out);
}
